// Attention_87565793230979
// MI455X (gfx1250) — compile-verified, matrix operand order checked by reading, not run on hardware
//
#include <hip/hip_runtime.h>
#include <hip/hip_bf16.h>

// Problem constants
#define B_     8
#define H_     32
#define W_     32
#define C_     384
#define NH_    6
#define HD_    64
#define HW_    1024          // H_*W_
#define NHEADS 48            // B_*NH_
#define M_     8192          // B_*HW_
#define SCALE_ 0.125f        // 1/sqrt(64)

typedef __attribute__((ext_vector_type(16))) __bf16 v16bf;
typedef __attribute__((ext_vector_type(8)))  float  v8f;

union FragU { uint4 u[2]; v16bf v; };

__device__ __forceinline__ unsigned short f2bf(float f) {
  unsigned int u = __float_as_uint(f);
  u += 0x7FFFu + ((u >> 16) & 1u);          // round-to-nearest-even
  return (unsigned short)(u >> 16);
}
__device__ __forceinline__ float bf2f(unsigned short b) {
  return __uint_as_float(((unsigned int)b) << 16);
}

// Load a 16x32 bf16 fragment (A-layout / mirrored B-layout).
// p points at the 32-element K-run for this lane's row/col; hi = lane>>4.
// lane<16 holds K {0..7,16..23}; lane>=16 holds K {8..15,24..31}.
__device__ __forceinline__ v16bf load_frag(const unsigned short* p, int hi) {
  FragU f;
  f.u[0] = *reinterpret_cast<const uint4*>(p + hi * 8);
  f.u[1] = *reinterpret_cast<const uint4*>(p + 16 + hi * 8);
  return f.v;
}

// ---------------------------------------------------------------------------
// Kernel 0: fp32 -> bf16 packing; weights transposed to [N][K] (K-contiguous)
// ---------------------------------------------------------------------------
__global__ __launch_bounds__(256) void pack_kernel(
    const float* __restrict__ x, const float* __restrict__ qkv_w,
    const float* __restrict__ proj_w,
    unsigned short* __restrict__ xb, unsigned short* __restrict__ wqkvt,
    unsigned short* __restrict__ pwt)
{
  const int XN = M_ * C_;          // 3,145,728
  const int WQ = C_ * 3 * C_;      //   442,368
  const int PW = C_ * C_;          //   147,456
  const int total = XN + WQ + PW;
  for (int i = blockIdx.x * blockDim.x + threadIdx.x; i < total;
       i += gridDim.x * blockDim.x) {
    if (i < XN) {
      xb[i] = f2bf(x[i]);
    } else if (i < XN + WQ) {
      int j = i - XN;
      int k = j / (3 * C_), n = j % (3 * C_);
      wqkvt[(size_t)n * C_ + k] = f2bf(qkv_w[j]);
    } else {
      int j = i - XN - WQ;
      int k = j / C_, n = j % C_;
      pwt[(size_t)n * C_ + k] = f2bf(proj_w[j]);
    }
  }
}

// ---------------------------------------------------------------------------
// WMMA GEMM: A[M][K] bf16 (K-contig) x Bt[N][K] bf16 (K-contig) -> fp32
// Block = 4 waves; each wave owns a 32(M) x 64(N) tile (2 M-frags x 4 N-frags,
// 8 WMMAs per 32-wide k-step, B-frags reused across both M-frags).
// Grid (N/64, M/128).
// MODE 0: out[m][n] = acc + bias[n]            (projection)
// MODE 1: scatter to q / k(prescaled) / v(transposed) bf16 buffers (QKV)
// ---------------------------------------------------------------------------
template <int MODE>
__global__ __launch_bounds__(128) void gemm_wmma(
    const unsigned short* __restrict__ A, const unsigned short* __restrict__ Bt,
    const float* __restrict__ bias, int M, int N, int K,
    float* __restrict__ outf,
    unsigned short* __restrict__ qb, unsigned short* __restrict__ kb,
    unsigned short* __restrict__ vt)
{
  const int lane = threadIdx.x & 31;
  const int wid  = threadIdx.x >> 5;
  const int r  = lane & 15;
  const int hi = lane >> 4;
  const int m0 = blockIdx.y * 128 + wid * 32;
  const int n0 = blockIdx.x * 64;

  v8f acc[2][4] = {};
  const unsigned short* arow0 = A + (size_t)(m0 + r) * K;
  const unsigned short* arow1 = A + (size_t)(m0 + 16 + r) * K;

  for (int kk = 0; kk < K; kk += 32) {
    v16bf a0 = load_frag(arow0 + kk, hi);
    v16bf a1 = load_frag(arow1 + kk, hi);
    if (kk + 32 < K) {
      __builtin_prefetch(arow0 + kk + 32, 0, 1);
      __builtin_prefetch(arow1 + kk + 32, 0, 1);
    }
#pragma unroll
    for (int t = 0; t < 4; ++t) {
      const unsigned short* bcol = Bt + (size_t)(n0 + t * 16 + r) * K + kk;
      v16bf b = load_frag(bcol, hi);
      acc[0][t] = __builtin_amdgcn_wmma_f32_16x16x32_bf16(
          false, a0, false, b, (short)0, acc[0][t], false, false);
      acc[1][t] = __builtin_amdgcn_wmma_f32_16x16x32_bf16(
          false, a1, false, b, (short)0, acc[1][t], false, false);
    }
  }

#pragma unroll
  for (int g = 0; g < 2; ++g) {
#pragma unroll
    for (int t = 0; t < 4; ++t) {
#pragma unroll
      for (int v = 0; v < 8; ++v) {
        const int row = m0 + g * 16 + v + 8 * hi;  // C-layout: M = v + 8*hi
        const int col = n0 + t * 16 + r;           //           N = lane&15
        float val = acc[g][t][v] + bias[col];
        if (MODE == 0) {
          outf[(size_t)row * N + col] = val;
        } else {
          const int which = col / C_;              // 0=q 1=k 2=v
          const int c  = col % C_;
          const int nh = c >> 6, hd = c & 63;
          const int bi = row >> 10, hw = row & 1023;
          const int bh = bi * NH_ + nh;
          if (which == 0)
            qb[(size_t)(bh * HW_ + hw) * HD_ + hd] = f2bf(val);
          else if (which == 1)
            kb[(size_t)(bh * HW_ + hw) * HD_ + hd] = f2bf(val * SCALE_);
          else
            vt[(size_t)(bh * HD_ + hd) * HW_ + hw] = f2bf(val);  // transposed
        }
      }
    }
  }
}

// ---------------------------------------------------------------------------
// Kernel 2: decomposed rel-pos biases.
// rel_h[bh][h][w][kh] = sum_c q[bh,h*32+w,c] * rel_pos_h[h-kh+31, c]
// rel_w[bh][h][w][kw] = sum_c q[bh,h*32+w,c] * rel_pos_w[w-kw+31, c]
// ---------------------------------------------------------------------------
__global__ __launch_bounds__(256) void rel_kernel(
    const unsigned short* __restrict__ qb, const float* __restrict__ rel_pos_h,
    const float* __restrict__ rel_pos_w,
    float* __restrict__ relh, float* __restrict__ relw)
{
  const int tid = blockIdx.x * blockDim.x + threadIdx.x;
  const int total = NHEADS * H_ * W_ * 32;
  if (tid >= total) return;
  const int k  = tid & 31;
  const int w  = (tid >> 5) & 31;
  const int h  = (tid >> 10) & 31;
  const int bh = tid >> 15;
  const unsigned short* q = qb + (size_t)(bh * HW_ + h * W_ + w) * HD_;
  const float* rh = rel_pos_h + (size_t)(h - k + (H_ - 1)) * HD_;
  const float* rw = rel_pos_w + (size_t)(w - k + (W_ - 1)) * HD_;
  float ah = 0.f, aw = 0.f;
#pragma unroll 8
  for (int c = 0; c < HD_; ++c) {
    float qc = bf2f(q[c]);
    ah += qc * rh[c];
    aw += qc * rw[c];
  }
  relh[tid] = ah;
  relw[tid] = aw;
}

// ---------------------------------------------------------------------------
// Kernel 3: flash attention. One wave per (head, pixel row of 32 queries).
// Queries hw in [qt*32, qt*32+32) all share h=qt, and local query index == w.
// A 32-key chunk is exactly one key row kh=kc, so the decomposed bias is
// rel_h[bh][h][w][kc] (per query) + rel_w[bh][h][w][kw] (per key column).
// Per chunk: 8 WMMAs for S (2 row-tiles x 2 col-tiles x K=64), online softmax,
// P through 2KB LDS (C-layout -> A-layout), 8 WMMAs for PV; K/V fragments are
// reused across both query row-tiles.
// ---------------------------------------------------------------------------
__global__ __launch_bounds__(32) void attn_flash(
    const unsigned short* __restrict__ qb, const unsigned short* __restrict__ kb,
    const unsigned short* __restrict__ vt, const float* __restrict__ relh,
    const float* __restrict__ relw, unsigned short* __restrict__ oa)
{
  __shared__ unsigned short Plds[32 * 32];   // 2 KB P tile staging
  const int lane = threadIdx.x;
  const int r  = lane & 15;
  const int hi = lane >> 4;
  const int qt = blockIdx.x;                 // 0..31: pixel row
  const int bh = blockIdx.y;                 // 0..47 head
  const int qbase = qt * 32;
  const int h = qt;

  // Q fragments: 2 row-tiles x 2 k-slices, loaded once
  v16bf qa[2][2];
#pragma unroll
  for (int g = 0; g < 2; ++g) {
    const unsigned short* qrow =
        qb + (size_t)(bh * HW_ + qbase + g * 16 + r) * HD_;
    qa[g][0] = load_frag(qrow, hi);
    qa[g][1] = load_frag(qrow + 32, hi);
  }

  float m[2][8], l[2][8];
#pragma unroll
  for (int g = 0; g < 2; ++g)
#pragma unroll
    for (int v = 0; v < 8; ++v) { m[g][v] = -1e30f; l[g][v] = 0.f; }
  v8f o[2][4] = {};
  const v8f vzero = {};

  const float* rhb = relh + (size_t)(bh * H_ + h) * W_ * 32;
  const float* rwb = relw + (size_t)(bh * H_ + h) * W_ * 32;

  for (int kc = 0; kc < 32; ++kc) {
    const int kbase = kc * 32;

    // K fragments (shared by both query row-tiles)
    v16bf kf[2][2];
#pragma unroll
    for (int t = 0; t < 2; ++t) {
      const unsigned short* kp =
          kb + (size_t)(bh * HW_ + kbase + t * 16 + r) * HD_;
      kf[t][0] = load_frag(kp, hi);
      kf[t][1] = load_frag(kp + 32, hi);
    }

    // S(32x32) = Qscaled . K^T  (scale folded into kb)
    v8f s[2][2];
#pragma unroll
    for (int g = 0; g < 2; ++g)
#pragma unroll
      for (int t = 0; t < 2; ++t) {
        s[g][t] = __builtin_amdgcn_wmma_f32_16x16x32_bf16(
            false, qa[g][0], false, kf[t][0], (short)0, vzero, false, false);
        s[g][t] = __builtin_amdgcn_wmma_f32_16x16x32_bf16(
            false, qa[g][1], false, kf[t][1], (short)0, s[g][t], false, false);
      }

    // rel bias + online softmax (row lives on 16 lanes of one half-wave)
#pragma unroll
    for (int g = 0; g < 2; ++g) {
#pragma unroll
      for (int v = 0; v < 8; ++v) {
        const int i  = v + 8 * hi;              // row within tile g
        const int wq = g * 16 + i;              // pixel col == local query idx
        const float rh = rhb[wq * 32 + kc];
        float x0 = s[g][0][v] + rh + rwb[wq * 32 + r];
        float x1 = s[g][1][v] + rh + rwb[wq * 32 + 16 + r];
        float rm = fmaxf(x0, x1);
#pragma unroll
        for (int off = 1; off < 16; off <<= 1)
          rm = fmaxf(rm, __shfl_xor(rm, off, 32));
        const float mn   = fmaxf(m[g][v], rm);
        const float corr = __expf(m[g][v] - mn);
        const float p0 = __expf(x0 - mn);
        const float p1 = __expf(x1 - mn);
        float rs = p0 + p1;
#pragma unroll
        for (int off = 1; off < 16; off <<= 1)
          rs += __shfl_xor(rs, off, 32);
        l[g][v] = l[g][v] * corr + rs;
        m[g][v] = mn;
#pragma unroll
        for (int t = 0; t < 4; ++t) o[g][t][v] *= corr;
        Plds[wq * 32 + r]      = f2bf(p0);
        Plds[wq * 32 + 16 + r] = f2bf(p1);
      }
    }
    asm volatile("s_wait_dscnt 0" ::: "memory");  // C-layout stores -> A-layout loads

    // Reload P as A-fragments (one 16x32 per row-tile)
    FragU pf[2];
#pragma unroll
    for (int g = 0; g < 2; ++g) {
      pf[g].u[0] = *reinterpret_cast<const uint4*>(&Plds[(g * 16 + r) * 32 + hi * 8]);
      pf[g].u[1] = *reinterpret_cast<const uint4*>(&Plds[(g * 16 + r) * 32 + 16 + hi * 8]);
    }
    asm volatile("s_wait_dscnt 0" ::: "memory");  // WAR guard vs next iter stores

    // O(32x64) += P(32x32) . V(32x64); vt is [head][hd][token] -> contiguous;
    // V fragments shared by both row-tiles.
#pragma unroll
    for (int t = 0; t < 4; ++t) {
      const unsigned short* vrow =
          vt + (size_t)(bh * HD_ + t * 16 + r) * HW_ + kbase;
      v16bf vf = load_frag(vrow, hi);
#pragma unroll
      for (int g = 0; g < 2; ++g)
        o[g][t] = __builtin_amdgcn_wmma_f32_16x16x32_bf16(
            false, pf[g].v, false, vf, (short)0, o[g][t], false, false);
    }
  }

  // normalize and scatter O (bf16) into [b][hw][nh*64+hd] for the proj GEMM
  const int b = bh / NH_, nh = bh % NH_;
#pragma unroll
  for (int g = 0; g < 2; ++g) {
#pragma unroll
    for (int v = 0; v < 8; ++v) {
      const float inv = 1.0f / l[g][v];
      const int hw = qbase + g * 16 + v + 8 * hi;
      const size_t base = (size_t)(b * HW_ + hw) * C_ + nh * HD_;
#pragma unroll
      for (int t = 0; t < 4; ++t)
        oa[base + t * 16 + r] = f2bf(o[g][t][v] * inv);
    }
  }
}

// ---------------------------------------------------------------------------
extern "C" void kernel_launch(void* const* d_in, const int* in_sizes, int n_in,
                              void* d_out, int out_size, void* d_ws,
                              size_t ws_size, hipStream_t stream)
{
  const float* x         = (const float*)d_in[0];
  const float* qkv_w     = (const float*)d_in[1];
  const float* qkv_b     = (const float*)d_in[2];
  const float* proj_w    = (const float*)d_in[3];
  const float* proj_b    = (const float*)d_in[4];
  const float* rel_pos_h = (const float*)d_in[5];
  const float* rel_pos_w = (const float*)d_in[6];
  float* out = (float*)d_out;

  char* ws = (char*)d_ws;
  size_t off = 0;
  auto alloc = [&](size_t bytes) -> void* {
    void* p = ws + off;
    off = (off + bytes + 255) & ~(size_t)255;
    return p;
  };
  unsigned short* xb    = (unsigned short*)alloc((size_t)M_ * C_ * 2);      // x bf16
  unsigned short* wqkvt = (unsigned short*)alloc((size_t)3 * C_ * C_ * 2);  // qkv_w^T
  unsigned short* pwt   = (unsigned short*)alloc((size_t)C_ * C_ * 2);      // proj_w^T
  unsigned short* qb    = (unsigned short*)alloc((size_t)NHEADS * HW_ * HD_ * 2);
  unsigned short* kb    = (unsigned short*)alloc((size_t)NHEADS * HW_ * HD_ * 2);
  unsigned short* vt    = (unsigned short*)alloc((size_t)NHEADS * HD_ * HW_ * 2);
  float* relh           = (float*)alloc((size_t)NHEADS * H_ * W_ * 32 * 4);
  float* relw           = (float*)alloc((size_t)NHEADS * H_ * W_ * 32 * 4);
  unsigned short* oa    = (unsigned short*)alloc((size_t)M_ * C_ * 2);      // attn out bf16
  (void)ws_size; (void)in_sizes; (void)n_in; (void)out_size;

  // 0) pack / convert / transpose
  pack_kernel<<<2048, 256, 0, stream>>>(x, qkv_w, proj_w, xb, wqkvt, pwt);

  // 1) QKV GEMM  [8192x384] x [384x1152], scatter q/k/v
  gemm_wmma<1><<<dim3(3 * C_ / 64, M_ / 128), 128, 0, stream>>>(
      xb, wqkvt, qkv_b, M_, 3 * C_, C_, nullptr, qb, kb, vt);

  // 2) rel-pos bias tables
  rel_kernel<<<(NHEADS * H_ * W_ * 32 + 255) / 256, 256, 0, stream>>>(
      qb, rel_pos_h, rel_pos_w, relh, relw);

  // 3) fused flash attention with decomposed rel-pos bias
  attn_flash<<<dim3(H_, NHEADS), 32, 0, stream>>>(
      qb, kb, vt, relh, relw, oa);

  // 4) projection GEMM  [8192x384] x [384x384] + bias -> fp32 out
  gemm_wmma<0><<<dim3(C_ / 64, M_ / 128), 128, 0, stream>>>(
      oa, pwt, proj_b, M_, C_, C_, out, nullptr, nullptr, nullptr);
}
